// cross_scale_trans_50792283242745
// MI455X (gfx1250) — matrix operand order repaired and checked by reading, not compiled, not run on hardware
//
#include <hip/hip_runtime.h>
#include <hip/hip_bf16.h>
#include <math.h>

// ---------------------------------------------------------------------------
// CDNA5 (gfx1250) wave32 WMMA types
// ---------------------------------------------------------------------------
typedef _Float16 half16 __attribute__((ext_vector_type(16)));
typedef _Float16 half2t __attribute__((ext_vector_type(2)));
typedef float    f32x8  __attribute__((ext_vector_type(8)));

#define DMODEL 256
#define HP     32     // N_HEADS * N_POINTS = 4*8 for every level
#define OFFD   96     // HP*3
#define GN_EPS 1e-5f

// ---------------------------------------------------------------------------
// Fragment helpers (wave32; A and B both use contiguous k = 16*(lane/16) + s)
// ---------------------------------------------------------------------------
// A matrix 16x32 f16 from LDS (row-major, stride lda elements, k-tile base k0)
__device__ __forceinline__ half16 frag_a_lds(const _Float16* A, int lda, int k0, int lane) {
  const int m = lane & 15, h = lane >> 4;
  half16 a;
#pragma unroll
  for (int j = 0; j < 8; ++j) {
    const int kb = 2 * j + 16 * h;
    half2t p = *(const half2t*)(A + m * lda + k0 + kb);   // even-aligned pair
    a[2 * j]     = p.x;
    a[2 * j + 1] = p.y;
  }
  return a;
}

// B matrix 32x16 f16 from a global row-major f32 weight [KROWS, ldb],
// k-tile base k0, n-tile base n0. Rows >= KROWS are zero-padded, but the
// loads stay UNCONDITIONAL (clamped address) so padding lowers to v_cndmask
// instead of serialized exec-mask branches; the predicate folds away entirely
// whenever KROWS is a multiple of 32.
template <int KROWS>
__device__ __forceinline__ half16 frag_b_gbl(const float* __restrict__ B, int ldb,
                                             int k0, int n0, int lane) {
  const int nn = lane & 15, h = lane >> 4;
  float raw[16];
#pragma unroll
  for (int j = 0; j < 16; ++j) {
    const int k = k0 + 16 * h + j;          // j covers both rows of each pair
    const int kc = (KROWS % 32 == 0) ? k : ((k < KROWS) ? k : 0);
    raw[j] = B[(size_t)kc * ldb + n0 + nn];
  }
  half16 b;
#pragma unroll
  for (int j = 0; j < 8; ++j) {
    const int ke = k0 + 16 * h + 2 * j;
    const bool v0 = (KROWS % 32 == 0) || (ke < KROWS);
    const bool v1 = (KROWS % 32 == 0) || (ke + 1 < KROWS);
    b[2 * j]     = v0 ? (_Float16)raw[2 * j]     : (_Float16)0.f;
    b[2 * j + 1] = v1 ? (_Float16)raw[2 * j + 1] : (_Float16)0.f;
  }
  return b;
}

// ---------------------------------------------------------------------------
// Stage 0: index-map fill + scatter  (lin -> voxel row, -1 = empty)
// ---------------------------------------------------------------------------
__global__ void fill_i32_kernel(int* __restrict__ p, int v, long count) {
  long i = (long)blockIdx.x * blockDim.x + threadIdx.x;
  long stride = (long)gridDim.x * blockDim.x;
  for (; i < count; i += stride) p[i] = v;
}

__global__ void scatter_map_kernel(const int* __restrict__ idx, int* __restrict__ map,
                                   int n, int H, int W) {
  int t = blockIdx.x * blockDim.x + threadIdx.x;
  if (t >= n) return;
  int z = idx[3 * t + 0], y = idx[3 * t + 1], x = idx[3 * t + 2];
  map[(long)z * H * W + (long)y * W + x] = t;
}

// ---------------------------------------------------------------------------
// Stage 1: fused projection per 16-voxel tile (block = 4 wave32s)
//   src = GroupNorm(feat @ Wi + bi)       [16, 256]   (LDS, f16)
//   q   = src @ Wq + bq                   [16, C]     (LDS, f16)
//   v   = src @ Wv + bv                   -> global   [n, C]
//   off = q @ Wso + bso                   -> global   [n, 96]
//   aw  = softmax(q @ Waw + baw, 32)      -> global   [n, 32]
// All matmuls via v_wmma_f32_16x16x32_f16.
// ---------------------------------------------------------------------------
template <int C>
__global__ __launch_bounds__(128) void proj_kernel(
    const float* __restrict__ feat,
    const float* __restrict__ Wi,  const float* __restrict__ bi,
    const float* __restrict__ gamma, const float* __restrict__ beta,
    const float* __restrict__ Wso, const float* __restrict__ bso,
    const float* __restrict__ Waw, const float* __restrict__ baw,
    const float* __restrict__ Wq,  const float* __restrict__ bq,
    const float* __restrict__ Wv,  const float* __restrict__ bv,
    float* __restrict__ v_out, float* __restrict__ off_out, float* __restrict__ aw_out,
    int n) {
  constexpr int KP  = (C <= 32) ? 32 : 64;   // padded K for C-sized contractions
  constexpr int NTC = C / 16;                // N-tiles of width 16 covering C

  const int tid  = threadIdx.x;
  const int wave = tid >> 5, lane = tid & 31;
  const int row0 = blockIdx.x * 16;

  __shared__ _Float16 sFeat[16][72];    // feat tile, zero-padded to KP
  __shared__ float    sY[16][DMODEL];   // pre-norm projection
  __shared__ _Float16 sSrc[16][264];    // normalized src (f16, padded stride)
  __shared__ _Float16 sQ[16][72];       // q tile (f16, zero-padded to KP)
  __shared__ float    sAw[16][HP];      // pre-softmax attention logits

  // ---- load feat tile (zero-padded rows/cols), zero q pad region ----
  for (int i = tid; i < 16 * 72; i += 128) {
    int r = i / 72, c = i - r * 72;
    float f = 0.f;
    if (c < C && (row0 + r) < n) f = feat[(size_t)(row0 + r) * C + c];
    sFeat[r][c] = (_Float16)f;
    sQ[r][c]    = (_Float16)0.f;
  }
  __syncthreads();

  // ---- GEMM 1: Y = feat @ Wi + bi   (K = C padded, N = 256) ----
  for (int nt = wave; nt < 16; nt += 4) {
    f32x8 acc = {};
#pragma unroll
    for (int kt = 0; kt < KP / 32; ++kt) {
      half16 a = frag_a_lds(&sFeat[0][0], 72, kt * 32, lane);
      half16 b = frag_b_gbl<C>(Wi, DMODEL, kt * 32, nt * 16, lane);
      acc = __builtin_amdgcn_wmma_f32_16x16x32_f16(false, a, false, b, (short)0, acc,
                                                   false, false);
    }
    const int nn = lane & 15, h = lane >> 4;
    const float bb = bi[nt * 16 + nn];
#pragma unroll
    for (int i = 0; i < 8; ++i) sY[8 * h + i][nt * 16 + nn] = acc[i] + bb;
  }
  __syncthreads();

  // ---- GroupNorm: groups = C, group size = 256/C ----
  constexpr int G = DMODEL / C;
  for (int job = tid; job < 16 * C; job += 128) {
    const int r = job / C, g = job - r * C;
    float mu = 0.f, m2 = 0.f;
#pragma unroll
    for (int e = 0; e < G; ++e) {
      float x = sY[r][g * G + e];
      mu += x; m2 += x * x;
    }
    mu *= (1.f / G);
    float var = m2 * (1.f / G) - mu * mu;
    float rs  = rsqrtf(var + GN_EPS);
#pragma unroll
    for (int e = 0; e < G; ++e) {
      const int c = g * G + e;
      sSrc[r][c] = (_Float16)((sY[r][c] - mu) * rs * gamma[c] + beta[c]);
    }
  }
  __syncthreads();

  // ---- GEMMs 2/3: q = src@Wq+bq (to LDS), v = src@Wv+bv (to global) ----
  for (int job = wave; job < 2 * NTC; job += 4) {
    const bool isQ = job < NTC;
    const int nt = isQ ? job : job - NTC;
    const float* W   = isQ ? Wq : Wv;
    const float* bia = isQ ? bq : bv;
    f32x8 acc = {};
#pragma unroll
    for (int kt = 0; kt < 8; ++kt) {
      half16 a = frag_a_lds(&sSrc[0][0], 264, kt * 32, lane);
      half16 b = frag_b_gbl<DMODEL>(W, C, kt * 32, nt * 16, lane);
      acc = __builtin_amdgcn_wmma_f32_16x16x32_f16(false, a, false, b, (short)0, acc,
                                                   false, false);
    }
    const int nn = lane & 15, h = lane >> 4;
    const float bb = bia[nt * 16 + nn];
    if (isQ) {
#pragma unroll
      for (int i = 0; i < 8; ++i) sQ[8 * h + i][nt * 16 + nn] = (_Float16)(acc[i] + bb);
    } else {
#pragma unroll
      for (int i = 0; i < 8; ++i) {
        const int r = 8 * h + i;
        if (row0 + r < n) v_out[(size_t)(row0 + r) * C + nt * 16 + nn] = acc[i] + bb;
      }
    }
  }
  __syncthreads();

  // ---- GEMMs 4/5: off = q@Wso+bso (6 tiles), awl = q@Waw+baw (2 tiles) ----
  for (int job = wave; job < 8; job += 4) {
    const bool isOff = job < 6;
    const int nt = isOff ? job : job - 6;
    const float* W   = isOff ? Wso : Waw;
    const float* bia = isOff ? bso : baw;
    const int ldb    = isOff ? OFFD : HP;
    f32x8 acc = {};
#pragma unroll
    for (int kt = 0; kt < KP / 32; ++kt) {
      half16 a = frag_a_lds(&sQ[0][0], 72, kt * 32, lane);
      half16 b = frag_b_gbl<C>(W, ldb, kt * 32, nt * 16, lane);
      acc = __builtin_amdgcn_wmma_f32_16x16x32_f16(false, a, false, b, (short)0, acc,
                                                   false, false);
    }
    const int nn = lane & 15, h = lane >> 4;
    const float bb = bia[nt * 16 + nn];
    if (isOff) {
#pragma unroll
      for (int i = 0; i < 8; ++i) {
        const int r = 8 * h + i;
        if (row0 + r < n) off_out[(size_t)(row0 + r) * OFFD + nt * 16 + nn] = acc[i] + bb;
      }
    } else {
#pragma unroll
      for (int i = 0; i < 8; ++i) sAw[8 * h + i][nt * 16 + nn] = acc[i] + bb;
    }
  }
  __syncthreads();

  // ---- softmax over 32 logits per row -> aw_out ----
  if (tid < 16 && (row0 + tid) < n) {
    float mx = -3.4e38f;
#pragma unroll
    for (int j = 0; j < HP; ++j) mx = fmaxf(mx, sAw[tid][j]);
    float s = 0.f;
#pragma unroll
    for (int j = 0; j < HP; ++j) s += __expf(sAw[tid][j] - mx);
    const float inv = 1.f / s;
#pragma unroll
    for (int j = 0; j < HP; ++j)
      aw_out[(size_t)(row0 + tid) * HP + j] = __expf(sAw[tid][j] - mx) * inv;
  }
}

// ---------------------------------------------------------------------------
// Stage 2: deformable trilinear sampling. One wave32 per voxel; lanes are
// channel-parallel (lane and lane+32 for C=64); gathers go through the compact
// index map + L2-resident v table instead of a dense value grid.
// ---------------------------------------------------------------------------
template <int C>
__global__ __launch_bounds__(256) void sample_kernel(
    const float* __restrict__ feat, const int* __restrict__ idx,
    const int* __restrict__ map, const float* __restrict__ v,
    const float* __restrict__ off, const float* __restrict__ aw,
    float* __restrict__ out, int n, int Dd, int Hh, int Ww) {
  __shared__ float sOff[8][OFFD];
  __shared__ float sAwv[8][HP];
  const int wave = threadIdx.x >> 5, lane = threadIdx.x & 31;
  const int r = blockIdx.x * 8 + wave;
  const bool active = r < n;

  if (active) {
#pragma unroll
    for (int j = 0; j < 3; ++j) sOff[wave][j * 32 + lane] = off[(size_t)r * OFFD + j * 32 + lane];
    sAwv[wave][lane] = aw[(size_t)r * HP + lane];
  }
  __syncthreads();
  if (!active) return;

  const int iz = idx[3 * r + 0], iy = idx[3 * r + 1], ix = idx[3 * r + 2];
  const float invD = 1.f / Dd, invH = 1.f / Hh, invW = 1.f / Ww;
  float acc0 = 0.f, acc1 = 0.f;

  for (int p = 0; p < HP; ++p) {
    const float awp = sAwv[wave][p];
    // reference convention: comp0 normalized by D mapped over W axis,
    // comp1 by H over H, comp2 by W over D (grid_sample x/y/z ordering).
    const float x  = (sOff[wave][3 * p + 0] + (float)iz) * invD * (float)Ww - 0.5f;
    const float y  = (sOff[wave][3 * p + 1] + (float)iy) * invH * (float)Hh - 0.5f;
    const float z  = (sOff[wave][3 * p + 2] + (float)ix) * invW * (float)Dd - 0.5f;
    const float xf = floorf(x), yf = floorf(y), zf = floorf(z);
    const float fx = x - xf, fy = y - yf, fz = z - zf;
    const int x0 = (int)xf, y0 = (int)yf, z0 = (int)zf;
#pragma unroll
    for (int dz = 0; dz < 2; ++dz) {
      const int zi = z0 + dz;
      const float wz = dz ? fz : 1.f - fz;
      const bool vz = (zi >= 0) && (zi < Dd);
      const int zc = min(max(zi, 0), Dd - 1);
#pragma unroll
      for (int dy = 0; dy < 2; ++dy) {
        const int yi = y0 + dy;
        const float wy = dy ? fy : 1.f - fy;
        const bool vy = vz && (yi >= 0) && (yi < Hh);
        const int yc = min(max(yi, 0), Hh - 1);
#pragma unroll
        for (int dx = 0; dx < 2; ++dx) {
          const int xi = x0 + dx;
          const float wx = dx ? fx : 1.f - fx;
          const float w = wz * wy * wx * awp;
          if (vy && (xi >= 0) && (xi < Ww) && w != 0.f) {
            const int xc = min(max(xi, 0), Ww - 1);
            const int rr = map[(long)zc * Hh * Ww + (long)yc * Ww + xc];
            if (rr >= 0) {
              const float* vp = v + (size_t)rr * C;
              if (lane < C) acc0 += w * vp[lane];
              if (C == 64)  acc1 += w * vp[lane + 32];
            }
          }
        }
      }
    }
  }
  if (lane < C) out[(size_t)r * C + lane] = feat[(size_t)r * C + lane] + acc0;
  if (C == 64)  out[(size_t)r * C + lane + 32] = feat[(size_t)r * C + lane + 32] + acc1;
}

// ---------------------------------------------------------------------------
// Host launcher
// ---------------------------------------------------------------------------
extern "C" void kernel_launch(void* const* d_in, const int* in_sizes, int n_in,
                              void* d_out, int out_size, void* d_ws, size_t ws_size,
                              hipStream_t stream) {
  (void)in_sizes; (void)n_in; (void)out_size; (void)ws_size;
  static const int NVOX[4] = {60000, 30000, 15000, 8000};
  static const int CH[4]   = {16, 32, 64, 64};
  static const int SP[4][3] = {{41, 400, 352}, {21, 200, 176}, {11, 100, 88}, {5, 50, 44}};

  char* ws = (char*)d_ws;
  size_t cur = 0;
  auto wsalloc = [&](size_t bytes) -> void* {
    void* p = ws + cur;
    cur = (cur + bytes + 255) & ~(size_t)255;
    return p;
  };

  size_t out_off = 0;
  for (int lvl = 0; lvl < 4; ++lvl) {
    const int n = NVOX[lvl], C = CH[lvl];
    const int Dd = SP[lvl][0], Hh = SP[lvl][1], Ww = SP[lvl][2];
    const long vox = (long)Dd * Hh * Ww;

    const int base = lvl * 14;  // feat, idx, then 12 params flat per level
    const float* feat  = (const float*)d_in[base + 0];
    const int*   idx   = (const int*)  d_in[base + 1];
    const float* Wi    = (const float*)d_in[base + 2];
    const float* bi    = (const float*)d_in[base + 3];
    const float* gamma = (const float*)d_in[base + 4];
    const float* beta  = (const float*)d_in[base + 5];
    const float* Wso   = (const float*)d_in[base + 6];
    const float* bso   = (const float*)d_in[base + 7];
    const float* Waw   = (const float*)d_in[base + 8];
    const float* baw   = (const float*)d_in[base + 9];
    const float* Wq    = (const float*)d_in[base + 10];
    const float* bq    = (const float*)d_in[base + 11];
    const float* Wv    = (const float*)d_in[base + 12];
    const float* bv    = (const float*)d_in[base + 13];

    int*   map  = (int*)  wsalloc((size_t)vox * 4);
    float* vbuf = (float*)wsalloc((size_t)n * C * 4);
    float* offb = (float*)wsalloc((size_t)n * OFFD * 4);
    float* awb  = (float*)wsalloc((size_t)n * HP * 4);
    float* out  = (float*)d_out + out_off;
    out_off += (size_t)n * C;

    // Stage 0: map init + scatter
    fill_i32_kernel<<<2048, 256, 0, stream>>>(map, -1, vox);
    scatter_map_kernel<<<(n + 255) / 256, 256, 0, stream>>>(idx, map, n, Hh, Ww);

    // Stage 1: fused WMMA projections
    dim3 pgrid((n + 15) / 16);
    if (C == 16) {
      proj_kernel<16><<<pgrid, 128, 0, stream>>>(feat, Wi, bi, gamma, beta, Wso, bso,
                                                 Waw, baw, Wq, bq, Wv, bv,
                                                 vbuf, offb, awb, n);
    } else if (C == 32) {
      proj_kernel<32><<<pgrid, 128, 0, stream>>>(feat, Wi, bi, gamma, beta, Wso, bso,
                                                 Waw, baw, Wq, bq, Wv, bv,
                                                 vbuf, offb, awb, n);
    } else {
      proj_kernel<64><<<pgrid, 128, 0, stream>>>(feat, Wi, bi, gamma, beta, Wso, bso,
                                                 Waw, baw, Wq, bq, Wv, bv,
                                                 vbuf, offb, awb, n);
    }

    // Stage 2: deformable sampling + residual
    dim3 sgrid((n + 7) / 8);
    if (C == 16) {
      sample_kernel<16><<<sgrid, 256, 0, stream>>>(feat, idx, map, vbuf, offb, awb,
                                                   out, n, Dd, Hh, Ww);
    } else if (C == 32) {
      sample_kernel<32><<<sgrid, 256, 0, stream>>>(feat, idx, map, vbuf, offb, awb,
                                                   out, n, Dd, Hh, Ww);
    } else {
      sample_kernel<64><<<sgrid, 256, 0, stream>>>(feat, idx, map, vbuf, offb, awb,
                                                   out, n, Dd, Hh, Ww);
    }
  }
}
